// MLA_73675868996045
// MI455X (gfx1250) — matrix operand order repaired and checked by reading, not compiled, not run on hardware
//
#include <hip/hip_runtime.h>

// ---------------- problem constants ----------------
#define B_    2
#define S_    2048
#define DIM_  2048
#define NH_   16
#define QL_   1536
#define KVL_  512
#define DN_   128
#define DR_   64
#define DV_   128
#define DQK_  192          // DN_ + DR_
#define DK_   576          // KVL_ + DR_  (attention contraction dim)
#define SCALE_F 0.07216878364870322f   // 192^-0.5

typedef unsigned short u16;
typedef float   v8f   __attribute__((ext_vector_type(8)));
typedef __bf16  v16bf __attribute__((ext_vector_type(16)));
typedef unsigned int u32x4 __attribute__((ext_vector_type(4)));
typedef int     i32x4 __attribute__((ext_vector_type(4)));

union Frag { v16bf v; u32x4 q[2]; };

__device__ __forceinline__ v16bf ldfrag(const u16* p0, const u16* p1) {
  Frag f;
  f.q[0] = *(const u32x4*)p0;
  f.q[1] = *(const u32x4*)p1;
  return f.v;
}

__device__ __forceinline__ v8f wmma_bf16(v16bf a, v16bf b, v8f c) {
  // D(f32 16x16) = A(16x32 bf16) * B(32x16 bf16) + C
  return __builtin_amdgcn_wmma_f32_16x16x32_bf16(false, a, false, b, (short)0, c, false, false);
}

__device__ __forceinline__ u16 f2bf(float f) {
  unsigned int u = __float_as_uint(f);
  u += 0x7fffu + ((u >> 16) & 1u);
  return (u16)(u >> 16);
}

__device__ __forceinline__ v8f v8zero() {
  v8f z = {0.f,0.f,0.f,0.f,0.f,0.f,0.f,0.f};
  return z;
}

// ---- CDNA5 async global->LDS copy (16B per lane), guarded with fallback ----
__device__ __forceinline__ void cp_b128(const u16* g, u16* l) {
#if defined(__gfx1250__) && __has_builtin(__builtin_amdgcn_global_load_async_to_lds_b128)
  __builtin_amdgcn_global_load_async_to_lds_b128(
      (__attribute__((address_space(1))) i32x4*)(void*)g,
      (__attribute__((address_space(3))) i32x4*)(void*)l,
      0, 0);
#else
  *(u32x4*)l = *(const u32x4*)g;
#endif
}
__device__ __forceinline__ void cp_wait() {
#if defined(__gfx1250__) && __has_builtin(__builtin_amdgcn_global_load_async_to_lds_b128) && \
    __has_builtin(__builtin_amdgcn_s_wait_asynccnt)
  __builtin_amdgcn_s_wait_asynccnt(0);
#endif
}

// ---------------------------------------------------------------------------
// Generic bf16 NT GEMM:  C[M,N] = A[M,K] * B[N,K]^T, f32 accumulate.
// Macro tile 128x128, BK=64 (16 WMMAs per barrier pair), 8 waves (4Mx2N),
// wave tile 32x64 (2x4 WMMA accs). z-batched via (z/zmod, z%zmod) strides.
// ---------------------------------------------------------------------------
__global__ __launch_bounds__(256) void gemm_bf16nt(
    const u16* __restrict__ A, const u16* __restrict__ B, void* __restrict__ C,
    int M, int N, int K, int lda, int ldb, int ldc, int out_bf16,
    int zmod, long sAd, long sAm, long sBd, long sBm, long sCd, long sCm)
{
  (void)M;
  const int z = blockIdx.z;
  A += (long)(z / zmod) * sAd + (long)(z % zmod) * sAm;
  B += (long)(z / zmod) * sBd + (long)(z % zmod) * sBm;
  const long coff = (long)(z / zmod) * sCd + (long)(z % zmod) * sCm;

  const int tid = threadIdx.x;
  const int w  = tid >> 5;
  const int l  = tid & 31;
  const int lg = l >> 4, lm = l & 15;
  const int wm = w & 3, wn = w >> 2;
  const int m0 = blockIdx.y * 128, n0 = blockIdx.x * 128;

  __shared__ u16 Al[128][72];   // 72-elem stride: 16B-aligned rows, conflict pad
  __shared__ u16 Bl[128][72];

  v8f acc[2][4];
#pragma unroll
  for (int mi = 0; mi < 2; ++mi)
#pragma unroll
    for (int ni = 0; ni < 4; ++ni) acc[mi][ni] = v8zero();

  const int arow = tid >> 1;          // 0..127
  const int acol = (tid & 1) * 32;    // 0 or 32

  for (int kt = 0; kt < K; kt += 64) {
    __syncthreads();
    {
      const u16* ap = A + (long)(m0 + arow) * lda + kt + acol;
#pragma unroll
      for (int i = 0; i < 4; ++i) cp_b128(ap + i*8, &Al[arow][acol + i*8]);
      __builtin_prefetch(ap + 64, 0, 0);           // global_prefetch_b8
      const int gn = n0 + arow;
      if (gn < N) {
        const u16* bp = B + (long)gn * ldb + kt + acol;
#pragma unroll
        for (int i = 0; i < 4; ++i) cp_b128(bp + i*8, &Bl[arow][acol + i*8]);
        __builtin_prefetch(bp + 64, 0, 0);
      } else {
        u32x4 zz = {0u,0u,0u,0u};
#pragma unroll
        for (int i = 0; i < 4; ++i) *(u32x4*)&Bl[arow][acol + i*8] = zz;
      }
    }
    cp_wait();
    __syncthreads();

#pragma unroll
    for (int sub = 0; sub < 2; ++sub) {
      v16bf afr[2], bfr[4];
#pragma unroll
      for (int mi = 0; mi < 2; ++mi) {
        // A lane layout: m = lane%16; k chunks {8*lg..}, {16+8*lg..}
        const u16* p = &Al[wm*32 + mi*16 + lm][sub*32 + 8*lg];
        afr[mi] = ldfrag(p, p + 16);
      }
#pragma unroll
      for (int ni = 0; ni < 4; ++ni) {
        // B lane layout: n = lane%16; k chunks {8*lg..}, {16+8*lg..}
        const u16* p = &Bl[wn*64 + ni*16 + lm][sub*32 + 8*lg];
        bfr[ni] = ldfrag(p, p + 16);
      }
#pragma unroll
      for (int mi = 0; mi < 2; ++mi)
#pragma unroll
        for (int ni = 0; ni < 4; ++ni)
          acc[mi][ni] = wmma_bf16(afr[mi], bfr[ni], acc[mi][ni]);
    }
  }

  // C/D layout: VGPR r -> row r + 8*lg, col = lane%16
#pragma unroll
  for (int mi = 0; mi < 2; ++mi) {
#pragma unroll
    for (int ni = 0; ni < 4; ++ni) {
      const int col = n0 + wn*64 + ni*16 + lm;
      if (col < N) {
#pragma unroll
        for (int r = 0; r < 8; ++r) {
          const int row = m0 + wm*32 + mi*16 + r + 8*lg;
          const float v = acc[mi][ni][r];
          if (out_bf16) ((u16*)C)[coff + (long)row*ldc + col] = f2bf(v);
          else          ((float*)C)[coff + (long)row*ldc + col] = v;
        }
      }
    }
  }
}

// ---------------------------------------------------------------------------
// Flash attention over the shared latent KV.
// block = 64 query rows of one (b,h); 8 waves.
//  scores: wave w computes tiles (mi = w>>1, ni = (w&1)*2 + {0,1}), K dim 576.
//  ctx:    wave w owns V columns [w*64, w*64+64), 4x4 WMMA accumulators.
// Q fragments (16 rows x 576) live in registers (18 v16bf).
// Whole 64x576 key tile staged in LDS once per key tile: 36 WMMAs / load.
// ---------------------------------------------------------------------------
__global__ __launch_bounds__(256) void mla_attn(
    const u16* __restrict__ Qfull,   // [B*NH][S][576]  (q_abs | rope(q_pe))
    const u16* __restrict__ Kmat,    // [B][S][576]     (kv_cache | rope(k_pe))
    const u16* __restrict__ Vt,      // [B][512][S]     transposed kv_cache
    u16* __restrict__ ctxA)          // [B*NH][S][512]
{
  const int z  = blockIdx.y;            // b*NH + h
  const int b  = z >> 4;
  const int qt = blockIdx.x;
  const int q0 = qt * 64;
  const int tid = threadIdx.x;
  const int w = tid >> 5, l = tid & 31;
  const int lg = l >> 4, lm = l & 15;
  const int mi = w >> 1;                // score tile row group 0..3
  const int npair = w & 1;              // score tile col pair
  const int cb = w * 64;                // ctx column block

  const u16* Qb = Qfull + ((long)z * S_ + q0) * DK_;
  const u16* Kb = Kmat  + (long)b * S_ * DK_;
  const u16* Vb = Vt    + (long)b * KVL_ * S_;

  __shared__ u16   Klds[64][DK_];       // 72 KB: full key tile
  __shared__ float Slds[64][64];
  __shared__ u16   Plds[64][64];
  __shared__ float m_s[64], l_s[64], alpha_s[64];
  __shared__ float red[64][4];

  // Q fragments in registers: rows mi*16 .. mi*16+15, all 576 k
  v16bf qf[18];
  {
    const u16* qrow = Qb + (long)(mi*16 + lm) * DK_;
#pragma unroll
    for (int kc = 0; kc < 18; ++kc) {
      const u16* p = qrow + kc*32 + 8*lg;
      qf[kc] = ldfrag(p, p + 16);
    }
  }

  v8f acc[4][4];
#pragma unroll
  for (int m2 = 0; m2 < 4; ++m2)
#pragma unroll
    for (int n2 = 0; n2 < 4; ++n2) acc[m2][n2] = v8zero();

  if (tid < 64) { m_s[tid] = -1e30f; l_s[tid] = 0.f; }
  __syncthreads();

  const int krow = tid >> 2;            // 0..63
  const int kqb  = (tid & 3) * 144;     // 4 threads cover one 576-elem row

  for (int kt = 0; kt <= qt; ++kt) {
    const int k0 = kt * 64;

    // ---- stage whole 64x576 key tile into LDS (async path on CDNA5)
    {
      const u16* kp = Kb + (long)(k0 + krow) * DK_ + kqb;
#pragma unroll
      for (int i = 0; i < 18; ++i) cp_b128(kp + i*8, &Klds[krow][kqb + i*8]);
    }
    cp_wait();
    __syncthreads();

    // ---- scores: S = Q . K^T over 576 dims (36 WMMAs, no barriers)
    v8f sacc[2];
    sacc[0] = v8zero(); sacc[1] = v8zero();
#pragma unroll
    for (int kch = 0; kch < 9; ++kch) {
#pragma unroll
      for (int sub = 0; sub < 2; ++sub) {
#pragma unroll
        for (int j = 0; j < 2; ++j) {
          const u16* p = &Klds[npair*32 + j*16 + lm][kch*64 + sub*32 + 8*lg];
          v16bf bf = ldfrag(p, p + 16);
          sacc[j] = wmma_bf16(qf[kch*2 + sub], bf, sacc[j]);
        }
      }
    }

    // ---- scale + causal mask, spill scores to LDS
#pragma unroll
    for (int j = 0; j < 2; ++j) {
      const int col = npair*32 + j*16 + lm;
#pragma unroll
      for (int r = 0; r < 8; ++r) {
        const int rowl = mi*16 + r + 8*lg;
        float v = sacc[j][r] * SCALE_F;
        if (kt == qt && col > rowl) v = -1e30f;
        Slds[rowl][col] = v;
      }
    }
    __syncthreads();

    // ---- online softmax (4 threads per row, 16 cols each)
    {
      const int row = tid >> 2, quad = tid & 3;
      float mx = -1e30f;
#pragma unroll
      for (int c = 0; c < 16; ++c) mx = fmaxf(mx, Slds[row][quad*16 + c]);
      red[row][quad] = mx;
      __syncthreads();
      if (quad == 0) {
        const float tm = fmaxf(fmaxf(red[row][0], red[row][1]),
                               fmaxf(red[row][2], red[row][3]));
        const float mold = m_s[row];
        const float mnew = fmaxf(mold, tm);
        alpha_s[row] = __expf(mold - mnew);
        m_s[row] = mnew;
      }
      __syncthreads();
      const float mrow = m_s[row];
      float sum = 0.f;
#pragma unroll
      for (int c = 0; c < 16; ++c) {
        const float pv = __expf(Slds[row][quad*16 + c] - mrow);
        Plds[row][quad*16 + c] = f2bf(pv);
        sum += pv;
      }
      red[row][quad] = sum;
      __syncthreads();
      if (quad == 0)
        l_s[row] = l_s[row] * alpha_s[row]
                 + red[row][0] + red[row][1] + red[row][2] + red[row][3];
      __syncthreads();
    }

    // ---- rescale running ctx by alpha(row)
#pragma unroll
    for (int m2 = 0; m2 < 4; ++m2) {
      float ar[8];
#pragma unroll
      for (int r = 0; r < 8; ++r) ar[r] = alpha_s[m2*16 + r + 8*lg];
#pragma unroll
      for (int n2 = 0; n2 < 4; ++n2)
#pragma unroll
        for (int r = 0; r < 8; ++r) acc[m2][n2][r] *= ar[r];
    }

    // ---- ctx += P . V   (V read k-major from pre-transposed Vt)
#pragma unroll
    for (int kc2 = 0; kc2 < 2; ++kc2) {
      v16bf pa[4];
#pragma unroll
      for (int m2 = 0; m2 < 4; ++m2) {
        const u16* p = &Plds[m2*16 + lm][kc2*32 + 8*lg];
        pa[m2] = ldfrag(p, p + 16);
      }
#pragma unroll
      for (int n2 = 0; n2 < 4; ++n2) {
        const u16* vp = Vb + (long)(cb + n2*16 + lm) * S_ + k0 + kc2*32 + 8*lg;
        v16bf vb = ldfrag(vp, vp + 16);
#pragma unroll
        for (int m2 = 0; m2 < 4; ++m2)
          acc[m2][n2] = wmma_bf16(pa[m2], vb, acc[m2][n2]);
      }
    }
    __syncthreads();   // all waves done with Klds/Plds before next stage
  }

  // ---- epilogue: ctx / l, store bf16
#pragma unroll
  for (int m2 = 0; m2 < 4; ++m2) {
    float li[8];
#pragma unroll
    for (int r = 0; r < 8; ++r) li[r] = 1.f / l_s[m2*16 + r + 8*lg];
#pragma unroll
    for (int n2 = 0; n2 < 4; ++n2) {
      const int cg = cb + n2*16 + lm;
#pragma unroll
      for (int r = 0; r < 8; ++r) {
        const int rg = q0 + m2*16 + r + 8*lg;
        ctxA[((long)z * S_ + rg) * KVL_ + cg] = f2bf(acc[m2][n2][r] * li[r]);
      }
    }
  }
}

// ---------------------------- prep kernels ---------------------------------
__global__ void cvt_f32_bf16(const float* __restrict__ in, u16* __restrict__ out, long n) {
  long i = (long)blockIdx.x * blockDim.x + threadIdx.x;
  const long st = (long)gridDim.x * blockDim.x;
  for (; i < n; i += st) out[i] = f2bf(in[i]);
}

// wukT[h][c][d] = wkv_b[(h*256 + d)*512 + c]   (transposed absorb weights)
__global__ void wuk_transpose(const float* __restrict__ wkv_b, u16* __restrict__ wukT, long n) {
  long i = (long)blockIdx.x * blockDim.x + threadIdx.x;
  const long st = (long)gridDim.x * blockDim.x;
  for (; i < n; i += st) {
    const int h = (int)(i >> 16);
    const int rem = (int)(i & 65535);
    const int c = rem >> 7, d = rem & 127;
    wukT[i] = f2bf(wkv_b[((long)(h*256 + d) << 9) + c]);
  }
}

__global__ __launch_bounds__(256) void rmsnorm_rows(
    const float* __restrict__ in, const float* __restrict__ w,
    u16* __restrict__ out, int len)
{
  const int row = blockIdx.x;
  const float* r = in + (long)row * len;
  __shared__ float red[256];
  float s = 0.f;
  for (int c = threadIdx.x; c < len; c += 256) { const float v = r[c]; s += v*v; }
  red[threadIdx.x] = s; __syncthreads();
  for (int off = 128; off > 0; off >>= 1) {
    if ((int)threadIdx.x < off) red[threadIdx.x] += red[threadIdx.x + off];
    __syncthreads();
  }
  const float rr = rsqrtf(red[0] / (float)len + 1e-6f);
  for (int c = threadIdx.x; c < len; c += 256)
    out[(long)row * len + c] = f2bf(r[c] * rr * w[c]);
}

// kv row: rmsnorm(first 512)*w -> Kmat[:, :512] and Vt (transposed); rope(k_pe) -> Kmat[:, 512:]
__global__ __launch_bounds__(256) void kv_prep(
    const float* __restrict__ kvfull, const float* __restrict__ w,
    const float* __restrict__ cosT, const float* __restrict__ sinT,
    u16* __restrict__ Kmat, u16* __restrict__ Vt)
{
  const int row = blockIdx.x;              // b*S + t
  const int bb = row >> 11, t = row & 2047;
  const float* r = kvfull + (long)row * DK_;
  __shared__ float red[256];
  float s = 0.f;
  for (int c = threadIdx.x; c < KVL_; c += 256) { const float v = r[c]; s += v*v; }
  red[threadIdx.x] = s; __syncthreads();
  for (int off = 128; off > 0; off >>= 1) {
    if ((int)threadIdx.x < off) red[threadIdx.x] += red[threadIdx.x + off];
    __syncthreads();
  }
  const float rr = rsqrtf(red[0] * (1.f / KVL_) + 1e-6f);
  for (int c = threadIdx.x; c < KVL_; c += 256) {
    const u16 u = f2bf(r[c] * rr * w[c]);
    Kmat[(long)row * DK_ + c] = u;
    Vt[((long)bb * KVL_ + c) * S_ + t] = u;
  }
  if (threadIdx.x < 32) {
    const int j = threadIdx.x;
    const float a = r[KVL_ + 2*j], bv = r[KVL_ + 2*j + 1];
    const float c = cosT[t*32 + j], sn = sinT[t*32 + j];
    Kmat[(long)row * DK_ + KVL_ + 2*j]     = f2bf(a*c - bv*sn);
    Kmat[(long)row * DK_ + KVL_ + 2*j + 1] = f2bf(a*sn + bv*c);
  }
}

// Qnope[z][s][d] = bf16(qproj[b][s][h*192 + d])
__global__ void qnope_pack(const float* __restrict__ qproj, u16* __restrict__ Qnope, long n) {
  long i = (long)blockIdx.x * blockDim.x + threadIdx.x;
  const long st = (long)gridDim.x * blockDim.x;
  for (; i < n; i += st) {
    const int d = (int)(i & 127);
    const long rest = i >> 7;
    const int s = (int)(rest & 2047);
    const int z = (int)(rest >> 11);
    const int bb = z >> 4, h = z & 15;
    Qnope[i] = f2bf(qproj[((long)bb * S_ + s) * (NH_*DQK_) + h*DQK_ + d]);
  }
}

// rope(q_pe) -> Qfull[z][s][512 + 2j(+1)]
__global__ void qpe_rope(const float* __restrict__ qproj,
                         const float* __restrict__ cosT, const float* __restrict__ sinT,
                         u16* __restrict__ Qfull, long n)
{
  long i = (long)blockIdx.x * blockDim.x + threadIdx.x;
  const long st = (long)gridDim.x * blockDim.x;
  for (; i < n; i += st) {
    const int j = (int)(i & 31);
    const long rest = i >> 5;
    const int s = (int)(rest & 2047);
    const int z = (int)(rest >> 11);
    const int bb = z >> 4, h = z & 15;
    const float* base = qproj + ((long)bb * S_ + s) * (NH_*DQK_) + h*DQK_ + DN_;
    const float a = base[2*j], bv = base[2*j + 1];
    const float c = cosT[s*32 + j], sn = sinT[s*32 + j];
    u16* o = Qfull + ((long)z * S_ + s) * DK_ + KVL_ + 2*j;
    o[0] = f2bf(a*c - bv*sn);
    o[1] = f2bf(a*sn + bv*c);
  }
}

// ---------------------------------------------------------------------------
extern "C" void kernel_launch(void* const* d_in, const int* in_sizes, int n_in,
                              void* d_out, int out_size, void* d_ws, size_t ws_size,
                              hipStream_t stream)
{
  (void)in_sizes; (void)n_in; (void)out_size; (void)ws_size;
  const float* x     = (const float*)d_in[0];
  const float* wq_a  = (const float*)d_in[1];
  const float* qnw   = (const float*)d_in[2];
  const float* wq_b  = (const float*)d_in[3];
  const float* wkv_a = (const float*)d_in[4];
  const float* kvnw  = (const float*)d_in[5];
  const float* wkv_b = (const float*)d_in[6];
  const float* wo    = (const float*)d_in[7];
  const float* cosT  = (const float*)d_in[8];
  const float* sinT  = (const float*)d_in[9];
  float* out = (float*)d_out;

  char* p = (char*)d_ws;
  auto alloc = [&](size_t bytes) -> char* {
    char* r = p;
    p += (bytes + 255) & ~(size_t)255;
    return r;
  };
  const long BS = (long)B_ * S_;   // 4096
  u16*  xb     = (u16*) alloc(BS * DIM_ * 2);
  u16*  wqab   = (u16*) alloc((long)QL_ * DIM_ * 2);
  u16*  wqbb   = (u16*) alloc((long)NH_*DQK_ * QL_ * 2);
  u16*  wkvab  = (u16*) alloc((long)DK_ * DIM_ * 2);
  u16*  wkvbb  = (u16*) alloc((long)NH_*(DN_+DV_) * KVL_ * 2);
  u16*  wukT   = (u16*) alloc((long)NH_ * KVL_ * DN_ * 2);
  u16*  wob    = (u16*) alloc((long)DIM_ * NH_*DV_ * 2);
  float* qlat  = (float*)alloc(BS * QL_ * 4);
  u16*  qlatn  = (u16*) alloc(BS * QL_ * 2);
  float* qproj = (float*)alloc(BS * (long)NH_*DQK_ * 4);
  float* kvfull= (float*)alloc(BS * DK_ * 4);
  u16*  Qnope  = (u16*) alloc((long)B_*NH_ * S_ * DN_ * 2);
  u16*  Qfull  = (u16*) alloc((long)B_*NH_ * S_ * DK_ * 2);
  u16*  Kmat   = (u16*) alloc((long)B_ * S_ * DK_ * 2);
  u16*  Vt     = (u16*) alloc((long)B_ * KVL_ * S_ * 2);
  u16*  ctxA   = (u16*) alloc((long)B_*NH_ * S_ * KVL_ * 2);
  u16*  headout= (u16*) alloc(BS * (long)NH_*DV_ * 2);

  const dim3 blk(256);

  // f32 -> bf16 conversions (weights + activations)
  cvt_f32_bf16<<<2048, blk, 0, stream>>>(x, xb, BS * DIM_);
  cvt_f32_bf16<<<2048, blk, 0, stream>>>(wq_a, wqab, (long)QL_ * DIM_);
  cvt_f32_bf16<<<2048, blk, 0, stream>>>(wq_b, wqbb, (long)NH_*DQK_ * QL_);
  cvt_f32_bf16<<<2048, blk, 0, stream>>>(wkv_a, wkvab, (long)DK_ * DIM_);
  cvt_f32_bf16<<<2048, blk, 0, stream>>>(wkv_b, wkvbb, (long)NH_*(DN_+DV_) * KVL_);
  cvt_f32_bf16<<<2048, blk, 0, stream>>>(wo, wob, (long)DIM_ * NH_*DV_);
  wuk_transpose<<<2048, blk, 0, stream>>>(wkv_b, wukT, (long)NH_ * KVL_ * DN_);

  // q_lat = x @ wq_a^T  (M=4096 N=1536 K=2048, f32 out)
  gemm_bf16nt<<<dim3(QL_/128, (int)(BS/128), 1), blk, 0, stream>>>(
      xb, wqab, qlat, (int)BS, QL_, DIM_, DIM_, DIM_, QL_, 0,
      1, 0,0, 0,0, 0,0);
  rmsnorm_rows<<<(int)BS, blk, 0, stream>>>(qlat, qnw, qlatn, QL_);

  // q = rms(q_lat) @ wq_b^T  (N=3072 K=1536, f32 out)
  gemm_bf16nt<<<dim3((NH_*DQK_)/128, (int)(BS/128), 1), blk, 0, stream>>>(
      qlatn, wqbb, qproj, (int)BS, NH_*DQK_, QL_, QL_, QL_, NH_*DQK_, 0,
      1, 0,0, 0,0, 0,0);

  // kv_full = x @ wkv_a^T  (N=576 K=2048, f32 out; N guarded)
  gemm_bf16nt<<<dim3((DK_+127)/128, (int)(BS/128), 1), blk, 0, stream>>>(
      xb, wkvab, kvfull, (int)BS, DK_, DIM_, DIM_, DIM_, DK_, 0,
      1, 0,0, 0,0, 0,0);

  qnope_pack<<<4096, blk, 0, stream>>>(qproj, Qnope, (long)B_*NH_*S_*DN_);
  qpe_rope<<<1024, blk, 0, stream>>>(qproj, cosT, sinT, Qfull, (long)B_*NH_*S_*(DR_/2));
  kv_prep<<<(int)BS, blk, 0, stream>>>(kvfull, kvnw, cosT, sinT, Kmat, Vt);

  // q_abs per (b,h): Qfull[:, :512] = Qnope @ wukT[h]^T  (M=2048 N=512 K=128, bf16 out, ldc=576)
  gemm_bf16nt<<<dim3(KVL_/128, S_/128, B_*NH_), blk, 0, stream>>>(
      Qnope, wukT, Qfull, S_, KVL_, DN_, DN_, DN_, DK_, 1,
      NH_,
      (long)NH_*S_*DN_, (long)S_*DN_,
      0L,               (long)KVL_*DN_,
      (long)NH_*S_*DK_, (long)S_*DK_);

  // flash attention over latent KV
  mla_attn<<<dim3(S_/64, B_*NH_), blk, 0, stream>>>(Qfull, Kmat, Vt, ctxA);

  // headout per (b,h): ctx @ W_UV^T  (M=2048 N=128 K=512, bf16 out into [bs][h*128+d])
  gemm_bf16nt<<<dim3(DV_/128, S_/128, B_*NH_), blk, 0, stream>>>(
      ctxA, wkvbb + (long)DN_*KVL_, headout, S_, DV_, KVL_, KVL_, KVL_, NH_*DV_, 1,
      NH_,
      (long)NH_*S_*KVL_, (long)S_*KVL_,
      0L,                (long)(DN_+DV_)*KVL_,
      (long)S_*NH_*DV_,  (long)DV_);

  // out = headout @ wo^T  (M=4096 N=2048 K=2048, f32 -> d_out)
  gemm_bf16nt<<<dim3((NH_*DV_)/128, (int)(BS/128), 1), blk, 0, stream>>>(
      headout, wob, out, (int)BS, NH_*DV_, NH_*DV_, NH_*DV_, NH_*DV_, DIM_, 0,
      1, 0,0, 0,0, 0,0);
}
